// CCTCFPWRsh_Net_83915071030124
// MI455X (gfx1250) — matrix operand order repaired and checked by reading, not compiled, not run on hardware
//
#include <hip/hip_runtime.h>
#include <hip/hip_bf16.h>

// ---------------------------------------------------------------------------
// Input index map (setup_inputs dict insertion order, params leaves in order,
// lists expanded element-by-element):
//  0:x 1:lat 2:lon 3:t 4:cc 5:tcf 6:pwr
//  7-10:lat_W[0..3] 11-14:lat_b[0..3] 15-18:lon_W[0..3] 19-22:lon_b[0..3]
//  23:c1W 24:c1b 25:c2W 26:c2b 27:c3W 28:c3b 29:c4W 30:c4b
//  31:ccW 32:ccb 33:tcfW 34:tcfb 35:pwrW 36:pwrb 37:tW 38:tb
//  39:gcnW 40:gcnb 41:gcnreW 42:gcnreb
// ---------------------------------------------------------------------------

typedef __attribute__((ext_vector_type(16))) _Float16 v16h;
typedef __attribute__((ext_vector_type(8)))  float    v8f;

// ---------------------------------------------------------------------------
// Tiny per-sample 4-layer tanh MLPs for lat / lon geo factors.
// ---------------------------------------------------------------------------
struct Mlp4 {
  const float *W0, *b0, *W1, *b1, *W2, *b2, *W3, *b3;
};

__global__ void mlp_kernel(const float* __restrict__ lat, const float* __restrict__ lon,
                           Mlp4 pl, Mlp4 pn,
                           float* __restrict__ latf, float* __restrict__ lonf, int B) {
  int tid = blockIdx.x * blockDim.x + threadIdx.x;
  if (tid >= 2 * B) return;
  int which = tid >= B;
  int b = which ? tid - B : tid;
  Mlp4 P = which ? pn : pl;
  const float* Ws[4] = {P.W0, P.W1, P.W2, P.W3};
  const float* bs[4] = {P.b0, P.b1, P.b2, P.b3};
  const int dims[5] = {1, 5, 25, 58, 156};

  float bufA[156], bufB[156];
  bufA[0] = (which ? lon : lat)[b];
  float* cur = bufA; float* nxt = bufB;
  int curn = 1;
  for (int l = 0; l < 4; ++l) {
    int on = dims[l + 1];
    for (int o = 0; o < on; ++o) {
      float s = bs[l][o];
      for (int i = 0; i < curn; ++i) s += Ws[l][o * curn + i] * cur[i];
      nxt[o] = tanhf(s);
    }
    float* tmp = cur; cur = nxt; nxt = tmp;
    curn = on;
  }
  float* dst = which ? lonf : latf;
  for (int j = 0; j < 156; ++j) dst[(size_t)b * 156 + j] = cur[j];
}

// ---------------------------------------------------------------------------
// Implicit-GEMM convolution with V_WMMA_F32_16X16X32_F16, geometry templated
// so all im2col address math strength-reduces to shifts/mads (no runtime
// integer division) and the K-loop trip count is a compile-time constant.
//
// One wave per 16(M outputs) x 16(N channels) tile; filters staged in LDS as
// f16, amortized over `mrep` consecutive M-tiles. Fused bias + ReLU.
// A-layout  (16x32 f16): lane L row M=L&15, K = (L>>4)*16 + e
//                        (e=2v+h), same contiguous k order as B
// B-layout  (32x16 f16): lane L col N=L&15, K = (L>>4)*16 + e  (e=2v+h)
//                        => 16 CONSECUTIVE f16 -> one 32B aligned LDS load
// C-layout  (16x16 f32): lane L col N=L&15, row M = (L>>4)*8 + r
// ---------------------------------------------------------------------------
template <int CinMem, int Cin, int H, int W, int Cout, int KH, int KW,
          int STRIDE, int PAD, int OH, int OW, int GEO>
__global__ __launch_bounds__(32)
void conv_wmma_kernel(const float* __restrict__ in, const float* __restrict__ wt,
                      const float* __restrict__ bias, float* __restrict__ out,
                      const float* __restrict__ latf, const float* __restrict__ lonf,
                      int mtiles, int mrep) {
  constexpr int M    = OH * OW;
  constexpr int KHW  = KH * KW;
  constexpr int K    = Cin * KHW;
  constexpr int Kpad = (K + 31) & ~31;

  __shared__ __align__(32) _Float16 wlds[16 * Kpad];

  const int lane  = threadIdx.x;
  const int b     = blockIdx.z;
  const int ntile = blockIdx.x;

  // Stage the 16 filters of this N-tile into LDS (f32 -> f16 once per block).
  for (int nn = 0; nn < 16; ++nn) {
    const int n = ntile * 16 + nn;
    const float* wrow = wt + (size_t)n * K;
    for (int k = lane; k < Kpad; k += 32)
      wlds[nn * Kpad + k] = (_Float16)((k < K && n < Cout) ? wrow[k] : 0.f);
  }
  __syncthreads();

  const int nloc = lane & 15;
  const int hi   = lane >> 4;
  const int n    = ntile * 16 + nloc;
  const float bv = (n < Cout) ? bias[n] : 0.f;
  const float* inb  = in + (size_t)b * CinMem * H * W;
  const float* latb = GEO ? (latf + (size_t)b * H) : nullptr;
  const float* lonb = GEO ? (lonf + (size_t)b * W) : nullptr;

  for (int mt = 0; mt < mrep; ++mt) {
    const int mtile = blockIdx.y * mrep + mt;
    if (mtile >= mtiles) break;

    const int  mg     = mtile * 16 + nloc;   // A row handled by this lane
    const bool mvalid = mg < M;
    const int  oy     = mvalid ? mg / OW : 0;   // constant divisor
    const int  ox     = mvalid ? mg % OW : 0;

    v8f acc;
#pragma unroll
    for (int r = 0; r < 8; ++r) acc[r] = bv;

#pragma unroll 2
    for (int k0 = 0; k0 < K; k0 += 32) {
      // ---- A fragment: im2col gather (zero for OOB / padded K / geo channel)
      v16h afrag;
#pragma unroll
      for (int e = 0; e < 16; ++e) {
        const int v = e >> 1, h = e & 1;
        const int k = k0 + hi * 16 + e; (void)v; (void)h;
        float val = 0.f;
        if (mvalid && k < K) {
          const int c   = k / KHW;          // constant divisors -> mul/shift
          const int rem = k - c * KHW;
          const int ky  = rem / KW;
          const int kx  = rem - ky * KW;
          const int iy  = oy * STRIDE + ky - PAD;
          const int ix  = ox * STRIDE + kx - PAD;
          if (iy >= 0 && iy < H && ix >= 0 && ix < W) {
            if (GEO && c == Cin - 1)
              val = latb[iy] * lonb[ix];
            else
              val = inb[(c * H + iy) * W + ix];
          }
        }
        afrag[e] = (_Float16)val;
      }

      // ---- B fragment: 16 consecutive f16 from zero-padded LDS filter row.
      //      element offset = nloc*Kpad + k0 + hi*16 (multiple of 16 -> 32B aligned)
      const v16h bfrag = *(const v16h*)&wlds[nloc * Kpad + k0 + hi * 16];

      acc = __builtin_amdgcn_wmma_f32_16x16x32_f16(false, afrag, false, bfrag,
                                                   (short)0, acc, false, false);
    }

    // ---- epilogue: bias already in acc; ReLU; scatter to NCHW
#pragma unroll
    for (int r = 0; r < 8; ++r) {
      const int m = mtile * 16 + hi * 8 + r;
      if (m < M && n < Cout) {
        float v = acc[r];
        v = v > 0.f ? v : 0.f;
        out[(((size_t)b * Cout + n) * OH + m / OW) * OW + (m % OW)] = v;
      }
    }
  }
}

// ---------------------------------------------------------------------------
// VALID max-pool with optional concat of a second source along channels.
// out row stride per batch is a parameter so the final pool can write straight
// into the fused [B,1176] output.
// ---------------------------------------------------------------------------
__global__ void relupool_kernel(const float* __restrict__ a, const float* __restrict__ b2,
                                int Ca, int Cb, int H, int W, int k, int s,
                                int OH, int OW, float* __restrict__ out,
                                long outRowStride, int B) {
  const int C = Ca + Cb;
  const long total = (long)B * C * OH * OW;
  for (long i = blockIdx.x * (long)blockDim.x + threadIdx.x; i < total;
       i += (long)gridDim.x * blockDim.x) {
    const int ox = (int)(i % OW);
    long r = i / OW;
    const int oy = (int)(r % OH); r /= OH;
    const int c  = (int)(r % C);
    const int b  = (int)(r / C);

    const float* src; int Csrc, cc2;
    if (c < Ca) { src = a;  Csrc = Ca; cc2 = c; }
    else        { src = b2; Csrc = Cb; cc2 = c - Ca; }
    const float* base = src + ((size_t)b * Csrc + cc2) * H * W;

    float m = -3.4e38f;
    for (int ky = 0; ky < k; ++ky)
      for (int kx = 0; kx < k; ++kx) {
        const float v = base[(size_t)(oy * s + ky) * W + (ox * s + kx)];
        m = v > m ? v : m;
      }
    out[(size_t)b * outRowStride + ((size_t)c * OH + oy) * OW + ox] = m;
  }
}

// ---------------------------------------------------------------------------
// Graph head: gf -> gf1 (identity-mask GCN reduces to per-node linear),
// batch-mean sigmoid adjacency, column-mean threshold, renormalized GCN,
// write 24 tail features per batch row. One block, B<=128.
// ---------------------------------------------------------------------------
__global__ __launch_bounds__(128)
void graph_kernel(const float* __restrict__ t, const float* __restrict__ cc,
                  const float* __restrict__ tcf, const float* __restrict__ pwr,
                  const float* __restrict__ ccW, const float* __restrict__ ccb,
                  const float* __restrict__ tcfW, const float* __restrict__ tcfb,
                  const float* __restrict__ pwrW, const float* __restrict__ pwrb,
                  const float* __restrict__ tW, const float* __restrict__ tb,
                  const float* __restrict__ gW, const float* __restrict__ gb,
                  const float* __restrict__ grW, const float* __restrict__ grb,
                  float* __restrict__ out, int B) {
  __shared__ float gf1s[128 * 24];
  __shared__ float adjS[36];
  __shared__ float AhatS[36];
  const int tid = threadIdx.x;

  if (tid < B) {
    float gf[6][4];
    const float c = cc[tid], tc = tcf[tid], pw = pwr[tid];
    for (int d = 0; d < 4; ++d) {
      gf[0][d] = c  * ccW[d]  + ccb[d];
      gf[1][d] = tc * tcfW[d] + tcfb[d];
      gf[2][d] = pw * pwrW[d] + pwrb[d];
    }
    for (int j = 0; j < 3; ++j)
      for (int d = 0; d < 4; ++d)
        gf[3 + j][d] = t[tid * 3 + j] * tW[d] + tb[d];
    // identity mask => Ahat == I, so gcn1 is a per-node linear layer
    for (int nn = 0; nn < 6; ++nn)
      for (int d = 0; d < 4; ++d) {
        float s = gb[d];
        for (int e = 0; e < 4; ++e) s += gf[nn][e] * gW[d * 4 + e];
        gf1s[tid * 24 + nn * 4 + d] = s;
      }
  }
  __syncthreads();

  if (tid < 36) {
    const int nn = tid / 6, mm = tid % 6;
    float s = 0.f;
    for (int b = 0; b < B; ++b) {
      float dot = 0.f;
      for (int d = 0; d < 4; ++d)
        dot += gf1s[b * 24 + nn * 4 + d] * gf1s[b * 24 + mm * 4 + d];
      s += 1.f / (1.f + expf(-dot));
    }
    adjS[tid] = s / (float)B;
  }
  __syncthreads();

  if (tid == 0) {
    float colmean[6];
    for (int mm = 0; mm < 6; ++mm) {
      float s = 0.f;
      for (int nn = 0; nn < 6; ++nn) s += adjS[nn * 6 + mm];
      colmean[mm] = s / 6.f;
    }
    float A2[6][6];
    for (int nn = 0; nn < 6; ++nn)
      for (int mm = 0; mm < 6; ++mm) {
        const float bn = adjS[nn * 6 + mm] > colmean[mm] ? 1.f : 0.f;
        const float bm = adjS[mm * 6 + nn] > colmean[nn] ? 1.f : 0.f;
        A2[nn][mm] = bn + bm + (nn == mm ? 1.f : 0.f);
      }
    float dinv[6];
    for (int nn = 0; nn < 6; ++nn) {
      float s = 0.f;
      for (int mm = 0; mm < 6; ++mm) s += A2[nn][mm];
      dinv[nn] = rsqrtf(s);
    }
    for (int nn = 0; nn < 6; ++nn)
      for (int mm = 0; mm < 6; ++mm)
        AhatS[nn * 6 + mm] = dinv[nn] * A2[nn][mm] * dinv[mm];
  }
  __syncthreads();

  if (tid < B) {
    float t1[6][4];
    for (int mm = 0; mm < 6; ++mm)
      for (int d = 0; d < 4; ++d) {
        float s = 0.f;
        for (int e = 0; e < 4; ++e) s += gf1s[tid * 24 + mm * 4 + e] * grW[d * 4 + e];
        t1[mm][d] = s;
      }
    for (int nn = 0; nn < 6; ++nn)
      for (int d = 0; d < 4; ++d) {
        float s = grb[d];
        for (int mm = 0; mm < 6; ++mm) s += AhatS[nn * 6 + mm] * t1[mm][d];
        out[(size_t)tid * 1176 + 1152 + nn * 4 + d] = s;
      }
  }
}

// ---------------------------------------------------------------------------
// Orchestration.
// ---------------------------------------------------------------------------
extern "C" void kernel_launch(void* const* d_in, const int* in_sizes, int n_in,
                              void* d_out, int out_size, void* d_ws, size_t ws_size,
                              hipStream_t stream) {
  (void)n_in; (void)out_size; (void)ws_size;
  const float* x   = (const float*)d_in[0];
  const float* lat = (const float*)d_in[1];
  const float* lon = (const float*)d_in[2];
  const float* tt  = (const float*)d_in[3];
  const float* cc  = (const float*)d_in[4];
  const float* tcf = (const float*)d_in[5];
  const float* pwr = (const float*)d_in[6];
  const int B = in_sizes[1];   // lat is [B,1]

  Mlp4 mlat = { (const float*)d_in[7],  (const float*)d_in[11],
                (const float*)d_in[8],  (const float*)d_in[12],
                (const float*)d_in[9],  (const float*)d_in[13],
                (const float*)d_in[10], (const float*)d_in[14] };
  Mlp4 mlon = { (const float*)d_in[15], (const float*)d_in[19],
                (const float*)d_in[16], (const float*)d_in[20],
                (const float*)d_in[17], (const float*)d_in[21],
                (const float*)d_in[18], (const float*)d_in[22] };

  const float* c1W = (const float*)d_in[23]; const float* c1b = (const float*)d_in[24];
  const float* c2W = (const float*)d_in[25]; const float* c2b = (const float*)d_in[26];
  const float* c3W = (const float*)d_in[27]; const float* c3b = (const float*)d_in[28];
  const float* c4W = (const float*)d_in[29]; const float* c4b = (const float*)d_in[30];

  float* ws = (float*)d_ws;
  float* latf   = ws;
  float* lonf   = latf + (size_t)B * 156;
  float* arenaA = lonf + (size_t)B * 156;            // max 16*50*50 floats / sample
  float* arenaB = arenaA + (size_t)B * 16 * 50 * 50; // pool buffers
  float* c1o = arenaA;
  float* p1o = arenaB;                                // [B,16,24,24]
  float* c2o = arenaA;                                // [B,32,20,20]
  float* p2o = arenaB;                                // r1 [B,32,9,9]
  float* c3o = arenaA;                                // [B,64,9,9]
  float* p3o = arenaB + (size_t)B * 32 * 9 * 9;       // r2 [B,96,4,4]
  float* c4o = arenaA;                                // [B,192,4,4]
  float* out = (float*)d_out;

  // 1) geo factor MLPs
  {
    int threads = 2 * B;
    mlp_kernel<<<(threads + 63) / 64, 64, 0, stream>>>(lat, lon, mlat, mlon, latf, lonf, B);
  }

  // 2) conv1: 9ch(8+geo) 7x7 s3 -> [B,16,50,50], M=2500 -> 157 tiles, mrep=8
  conv_wmma_kernel<8, 9, 156, 156, 16, 7, 7, 3, 0, 50, 50, 1>
      <<<dim3(1, 20, B), 32, 0, stream>>>(x, c1W, c1b, c1o, latf, lonf, 157, 8);
  relupool_kernel<<<dim3(2048), 256, 0, stream>>>(c1o, nullptr, 16, 0, 50, 50, 3, 2, 24, 24,
                                                  p1o, (long)16 * 24 * 24, B);

  // 3) conv2: 16->32 5x5 -> [B,32,20,20], M=400 -> 25 tiles, mrep=4
  conv_wmma_kernel<16, 16, 24, 24, 32, 5, 5, 1, 0, 20, 20, 0>
      <<<dim3(2, 7, B), 32, 0, stream>>>(p1o, c2W, c2b, c2o, nullptr, nullptr, 25, 4);
  relupool_kernel<<<dim3(1024), 256, 0, stream>>>(c2o, nullptr, 32, 0, 20, 20, 3, 2, 9, 9,
                                                  p2o, (long)32 * 9 * 9, B);

  // 4) conv3: 32->64 3x3 p1 -> [B,64,9,9], M=81 -> 6 tiles, mrep=2
  conv_wmma_kernel<32, 32, 9, 9, 64, 3, 3, 1, 1, 9, 9, 0>
      <<<dim3(4, 3, B), 32, 0, stream>>>(p2o, c3W, c3b, c3o, nullptr, nullptr, 6, 2);
  // concat(r1, conv3) + pool3 -> r2 [B,96,4,4]
  relupool_kernel<<<dim3(512), 256, 0, stream>>>(p2o, c3o, 32, 64, 9, 9, 3, 2, 4, 4,
                                                 p3o, (long)96 * 4 * 4, B);

  // 5) conv4: 96->192 3x3 p1 -> [B,192,4,4], M=16 -> 1 tile
  conv_wmma_kernel<96, 96, 4, 4, 192, 3, 3, 1, 1, 4, 4, 0>
      <<<dim3(12, 1, B), 32, 0, stream>>>(p3o, c4W, c4b, c4o, nullptr, nullptr, 1, 1);
  // concat(r2, conv4) + pool4 -> straight into fused output rows [B,1176]
  relupool_kernel<<<dim3(512), 256, 0, stream>>>(p3o, c4o, 96, 192, 4, 4, 2, 2, 2, 2,
                                                 out, 1176L, B);

  // 6) graph head -> out[:, 1152:1176]
  graph_kernel<<<1, 128, 0, stream>>>(
      tt, cc, tcf, pwr,
      (const float*)d_in[31], (const float*)d_in[32],
      (const float*)d_in[33], (const float*)d_in[34],
      (const float*)d_in[35], (const float*)d_in[36],
      (const float*)d_in[37], (const float*)d_in[38],
      (const float*)d_in[39], (const float*)d_in[40],
      (const float*)d_in[41], (const float*)d_in[42],
      out, B);
}
